// TranslitModel_69054484185817
// MI455X (gfx1250) — matrix operand order repaired and checked by reading, not compiled, not run on hardware
//
#include <hip/hip_runtime.h>
#include <hip/hip_bf16.h>
#include <stdint.h>

// ---------------------------------------------------------------------------
// Seq2seq GRU (2-layer biGRU encoder + 4-cell decoder + classifier + NLL loss)
// for MI455X / gfx1250.
//
//  * All GEMMs run on V_WMMA_F32_16X16X32_BF16 (bf16 in, fp32 accum).
//  * GEMM staging uses GLOBAL_LOAD_ASYNC_TO_LDS_B128 (ASYNCcnt-tracked direct
//    global->LDS copy) with double-buffered LDS so the copy of K-slab k+1
//    overlaps the WMMA work on slab k; one s_wait_asynccnt + one barrier per
//    slab.  Falls back to synchronous staging if the builtins are absent.
//  * Per time step, one fused GEMM kernel computes gi = x_t @ Wih^T and
//    gh = h @ Whh^T for BOTH directions (gridDim.z) into a [2,B,3072] scratch;
//    a small elementwise kernel then does the r/z/n gate math in fp32.
//  * Encoder finals live in the same h slots the decoder starts from.
//  * Classifier: one WMMA GEMM [T*B,1024]x[1024,500], bias + t-major->b-major
//    row remap fused in the epilogue.  Loss/acc: per-row logsumexp+argmax,
//    then a single-block reduction.
// ---------------------------------------------------------------------------

typedef __attribute__((ext_vector_type(16))) __bf16 v16bf;
typedef __attribute__((ext_vector_type(8)))  __bf16 v8bf;
typedef __attribute__((ext_vector_type(8)))  float  v8f;
typedef int v4i __attribute__((vector_size(16)));   // builtin's pointee type

#if defined(__has_builtin)
#if __has_builtin(__builtin_amdgcn_global_load_async_to_lds_b128) && \
    __has_builtin(__builtin_amdgcn_s_wait_asynccnt)
#define USE_ASYNC_LDS 1
#endif
#endif
#ifndef USE_ASYNC_LDS
#define USE_ASYNC_LDS 0
#endif

#define AS1 __attribute__((address_space(1)))
#define AS3 __attribute__((address_space(3)))

namespace {
constexpr int kB  = 1024;   // batch
constexpr int kT  = 64;     // seq len
constexpr int kE  = 256;    // embedding
constexpr int kH  = 512;    // hidden
constexpr int kH3 = 1536;   // 3*H
constexpr int kV  = 500;    // vocab (target)
constexpr int k2H = 1024;   // 2*H
constexpr int kG  = 3072;   // gi(1536) | gh(1536) per batch row
constexpr int kSlab = 128 * 32;  // one LDS tile slab (elements)
}

__device__ __forceinline__ v16bf cat8(v8bf lo, v8bf hi) {
  return __builtin_shufflevector(lo, hi, 0,1,2,3,4,5,6,7,8,9,10,11,12,13,14,15);
}

// ---------------------------------------------------------------------------
// Stage one 128x32 A slab + one 128x32 B slab into LDS.
// Async path: per-lane GLOBAL_LOAD_ASYNC_TO_LDS_B128 (no VGPR round trip).
// ---------------------------------------------------------------------------
__device__ __forceinline__ void stage_slab(
    const __bf16* __restrict__ A, int lda,
    const __bf16* __restrict__ Bw, int ldb, int nValid,
    int mBlock, int nBlock, int k0,
    __bf16* dA, __bf16* dB)
{
  const int tid = threadIdx.x;
#if USE_ASYNC_LDS
  #pragma unroll
  for (int c = tid; c < 512; c += 256) {
    const int r = c >> 2, cc = (c & 3) << 3;
    __builtin_amdgcn_global_load_async_to_lds_b128(
        (AS1 v4i*)(uintptr_t)(A + (size_t)(mBlock + r) * lda + k0 + cc),
        (AS3 v4i*)(uint32_t)(uintptr_t)(dA + r * 32 + cc), 0, 0);
  }
  #pragma unroll
  for (int c = tid; c < 512; c += 256) {
    const int r = c >> 2, cc = (c & 3) << 3;
    if (nBlock + r < nValid) {
      __builtin_amdgcn_global_load_async_to_lds_b128(
          (AS1 v4i*)(uintptr_t)(Bw + (size_t)(nBlock + r) * ldb + k0 + cc),
          (AS3 v4i*)(uint32_t)(uintptr_t)(dB + r * 32 + cc), 0, 0);
    } else {
      v8bf z;
      #pragma unroll
      for (int q = 0; q < 8; ++q) z[q] = (__bf16)0.0f;
      *(v8bf*)(dB + r * 32 + cc) = z;
    }
  }
#else
  #pragma unroll
  for (int c = tid; c < 512; c += 256) {
    const int r = c >> 2, cc = (c & 3) << 3;
    v8bf va = *(const v8bf*)(A + (size_t)(mBlock + r) * lda + k0 + cc);
    *(v8bf*)(dA + r * 32 + cc) = va;
  }
  #pragma unroll
  for (int c = tid; c < 512; c += 256) {
    const int r = c >> 2, cc = (c & 3) << 3;
    v8bf vb;
    if (nBlock + r < nValid) {
      vb = *(const v8bf*)(Bw + (size_t)(nBlock + r) * ldb + k0 + cc);
    } else {
      #pragma unroll
      for (int q = 0; q < 8; ++q) vb[q] = (__bf16)0.0f;
    }
    *(v8bf*)(dB + r * 32 + cc) = vb;
  }
#endif
}

// One K-slab of WMMA work: 2x4 fragments of v_wmma_f32_16x16x32_bf16.
// Fragment layouts per ISA 7.12.2 (A and B: lanes<16 K{0-7,16-23}, lanes>=16
// K{8-15,24-31}).
__device__ __forceinline__ void compute_slab(
    const __bf16* cA, const __bf16* cB,
    int wm, int wn, int lrow, int lhi, v8f acc[2][4])
{
  v16bf af[2], bwf[4];
  #pragma unroll
  for (int i = 0; i < 2; ++i) {
    const __bf16* p = cA + (wm * 32 + i * 16 + lrow) * 32;
    af[i] = cat8(*(const v8bf*)(p + lhi * 8), *(const v8bf*)(p + lhi * 8 + 16));
  }
  #pragma unroll
  for (int j = 0; j < 4; ++j) {
    const __bf16* p = cB + (wn * 64 + j * 16 + lrow) * 32 + lhi * 8;
    bwf[j] = cat8(*(const v8bf*)(p), *(const v8bf*)(p + 16));
  }
  #pragma unroll
  for (int i = 0; i < 2; ++i)
    #pragma unroll
    for (int j = 0; j < 4; ++j)
      acc[i][j] = __builtin_amdgcn_wmma_f32_16x16x32_bf16(
          false, af[i], false, bwf[j], (short)0, acc[i][j], false, false);
}

// ---------------------------------------------------------------------------
// 128x128 block tile GEMM core: C[128,128] += A[128xK] * Bw[NxK]^T
// (both K-major bf16 rows).  8 wave32 waves, each owns a 32x64 sub-tile.
// Double-buffered LDS; one barrier + (async path) one s_wait_asynccnt per
// K-slab.  A rows must exist for the whole 128-row block (all our Ms are
// multiples of 128); Bw rows guarded against nValid (classifier N=500).
// lsA/lsB each hold 2 slabs (2 * 128*32 elements).
// ---------------------------------------------------------------------------
__device__ __forceinline__ void wmma_tile_128x128(
    const __bf16* __restrict__ A, int lda,
    const __bf16* __restrict__ Bw, int ldb, int nValid,
    int K, int mBlock, int nBlock,
    __bf16* lsA, __bf16* lsB, v8f acc[2][4])
{
  const int tid  = threadIdx.x;
  const int wave = tid >> 5, lane = tid & 31;
  const int wm = wave & 3, wn = wave >> 2;
  const int lrow = lane & 15, lhi = lane >> 4;

  #pragma unroll
  for (int i = 0; i < 2; ++i)
    #pragma unroll
    for (int j = 0; j < 4; ++j)
      #pragma unroll
      for (int r = 0; r < 8; ++r) acc[i][j][r] = 0.0f;

  stage_slab(A, lda, Bw, ldb, nValid, mBlock, nBlock, 0, lsA, lsB);

  for (int k0 = 0; k0 < K; k0 += 32) {
    const int cur = (k0 >> 5) & 1;
#if USE_ASYNC_LDS
    __builtin_amdgcn_s_wait_asynccnt(0);   // our slab-k copies have landed
#endif
    __syncthreads();                        // everyone's copies visible; all
                                            // reads of buf[1-cur] retired
    if (k0 + 32 < K) {
      stage_slab(A, lda, Bw, ldb, nValid, mBlock, nBlock, k0 + 32,
                 lsA + (1 - cur) * kSlab, lsB + (1 - cur) * kSlab);
      // Prefetch the slab after next into L2 while we are at it.
      if (k0 + 64 < K)
        __builtin_prefetch(A + (size_t)(mBlock + (tid >> 1)) * lda + (k0 + 64), 0, 1);
    }
    compute_slab(lsA + cur * kSlab, lsB + cur * kSlab, wm, wn, lrow, lhi, acc);
  }
}

// ---------------------------------------------------------------------------
// Per-time-step fused GEMM: gridDim = (24, 8, 2).
//   blockIdx.x < 12  -> gi half : A = x_t      [B,Kin], W = Wih [1536,Kin]
//   blockIdx.x >= 12 -> gh half : A = h (bf16) [B,512], W = Whh [1536,512]
//   blockIdx.z = direction (fwd / "bwd" cell).
// Writes g[dir][b][0:1536]=gi, g[dir][b][1536:3072]=gh (fp32, no bias).
// ---------------------------------------------------------------------------
__global__ __launch_bounds__(256) void gru_step_gemm(
    const __bf16* __restrict__ xbuf, int Kin,
    const __bf16* __restrict__ wihF, const __bf16* __restrict__ wihB,
    const __bf16* __restrict__ whhF, const __bf16* __restrict__ whhB,
    const __bf16* __restrict__ hbF,  const __bf16* __restrict__ hbB,
    float* __restrict__ g, int tF, int tB)
{
  __shared__ __bf16 lsA[2 * kSlab];
  __shared__ __bf16 lsB[2 * kSlab];
  const int dir    = blockIdx.z;
  const int half   = blockIdx.x / 12;
  const int nBlock = (blockIdx.x % 12) * 128;
  const int mBlock = blockIdx.y * 128;

  const __bf16* A; const __bf16* W; int K;
  if (half == 0) {
    const int t = dir ? tB : tF;
    A = xbuf + (size_t)t * kB * Kin; K = Kin;
    W = dir ? wihB : wihF;
  } else {
    A = dir ? hbB : hbF; K = kH;
    W = dir ? whhB : whhF;
  }
  v8f acc[2][4];
  wmma_tile_128x128(A, K, W, K, kH3, K, mBlock, nBlock, lsA, lsB, acc);

  float* Cb = g + (size_t)dir * (kB * kG) + half * kH3;
  const int wave = threadIdx.x >> 5, lane = threadIdx.x & 31;
  const int wm = wave & 3, wn = wave >> 2, lrow = lane & 15, lhi = lane >> 4;
  #pragma unroll
  for (int i = 0; i < 2; ++i)
    #pragma unroll
    for (int j = 0; j < 4; ++j)
      #pragma unroll
      for (int r = 0; r < 8; ++r) {
        const int row = mBlock + wm * 32 + i * 16 + lhi * 8 + r;  // batch index
        const int col = nBlock + wn * 64 + j * 16 + lrow;         // gate index
        Cb[(size_t)row * kG + col] = acc[i][j][r];
      }
}

// ---------------------------------------------------------------------------
// GRU gate math (fp32): r,z,n + state update; writes h (fp32), bf16 mirror,
// and optionally y[t] (bf16, [T,B,2H] time-major, dir selects column half).
// ---------------------------------------------------------------------------
__global__ __launch_bounds__(256) void gru_gate(
    const float* __restrict__ g,
    const float* __restrict__ bihF, const float* __restrict__ bhhF,
    const float* __restrict__ bihB, const float* __restrict__ bhhB,
    float* __restrict__ hF, float* __restrict__ hB,
    __bf16* __restrict__ hbF, __bf16* __restrict__ hbB,
    __bf16* __restrict__ y, int tF, int tB)
{
  const int idx = blockIdx.x * 256 + threadIdx.x;   // < 2*B*H
  const int dir = idx >= kB * kH;
  const int e   = idx - dir * kB * kH;
  const int b   = e >> 9;            // H = 512
  const int j   = e & (kH - 1);
  const float* gg = g + (size_t)dir * (kB * kG) + (size_t)b * kG;
  const float* bi = dir ? bihB : bihF;
  const float* bh = dir ? bhhB : bhhF;

  const float ir = gg[j]              + bi[j];
  const float iz = gg[kH + j]         + bi[kH + j];
  const float in = gg[2 * kH + j]     + bi[2 * kH + j];
  const float hr = gg[kH3 + j]        + bh[j];
  const float hz = gg[kH3 + kH + j]   + bh[kH + j];
  const float hn = gg[kH3 + 2*kH + j] + bh[2 * kH + j];

  const float r = 1.0f / (1.0f + __expf(-(ir + hr)));
  const float z = 1.0f / (1.0f + __expf(-(iz + hz)));
  const float n = tanhf(in + r * hn);

  float* h = dir ? hB : hF;
  const float hnew = (1.0f - z) * n + z * h[e];
  h[e] = hnew;
  (dir ? hbB : hbF)[e] = (__bf16)hnew;
  if (y) {
    const int t = dir ? tB : tF;
    y[(size_t)t * (kB * k2H) + (size_t)b * k2H + dir * kH + j] = (__bf16)hnew;
  }
}

// ---------------------------------------------------------------------------
// Classifier: logits = y_dec1[T*B,1024] @ cls_W[500,1024]^T + b; remap the
// time-major row t*B+b to output row b*T+t.  grid = (4, 512).
// ---------------------------------------------------------------------------
__global__ __launch_bounds__(256) void cls_gemm(
    const __bf16* __restrict__ yin, const __bf16* __restrict__ W,
    const float* __restrict__ bias, float* __restrict__ out)
{
  __shared__ __bf16 lsA[2 * kSlab];
  __shared__ __bf16 lsB[2 * kSlab];
  const int nBlock = blockIdx.x * 128;
  const int mBlock = blockIdx.y * 128;
  v8f acc[2][4];
  wmma_tile_128x128(yin, k2H, W, k2H, kV, k2H, mBlock, nBlock, lsA, lsB, acc);

  const int wave = threadIdx.x >> 5, lane = threadIdx.x & 31;
  const int wm = wave & 3, wn = wave >> 2, lrow = lane & 15, lhi = lane >> 4;
  #pragma unroll
  for (int i = 0; i < 2; ++i)
    #pragma unroll
    for (int j = 0; j < 4; ++j)
      #pragma unroll
      for (int r = 0; r < 8; ++r) {
        const int row = mBlock + wm * 32 + i * 16 + lhi * 8 + r;  // = t*B + b
        const int col = nBlock + wn * 64 + j * 16 + lrow;
        if (col < kV) {
          const int t = row >> 10, b = row & (kB - 1);
          out[((size_t)b * kT + t) * kV + col] = acc[i][j][r] + bias[col];
        }
      }
}

// ---------------------------------------------------------------------------
// Utilities
// ---------------------------------------------------------------------------
__global__ void f32_to_bf16_k(const float* __restrict__ s, __bf16* __restrict__ d, int n) {
  const int i = blockIdx.x * 256 + threadIdx.x;
  if (i < n) d[i] = (__bf16)s[i];
}

__global__ void embed_enc(const float* __restrict__ emb, const int* __restrict__ src,
                          __bf16* __restrict__ xb) {
  const int i = blockIdx.x * 256 + threadIdx.x;       // T*B*E total (exact)
  const int e = i & (kE - 1); const int rb = i >> 8;  // row = t*B + b
  const int b = rb & (kB - 1); const int t = rb >> 10;
  const int tok = src[b * kT + t];
  xb[i] = (__bf16)emb[tok * kE + e];
}

__global__ void embed_dec(const float* __restrict__ emb, const int* __restrict__ tgt,
                          __bf16* __restrict__ xb) {
  const int i = blockIdx.x * 256 + threadIdx.x;
  const int e = i & (kE - 1); const int rb = i >> 8;
  const int b = rb & (kB - 1); const int t = rb >> 10;
  const int tok = (t == 0) ? 0 : tgt[b * kT + t - 1];  // SOS token = 0
  const float v = emb[tok * kE + e];
  xb[i] = (__bf16)(v > 0.0f ? v : 0.0f);               // fused relu
}

__global__ void zero_state(float* __restrict__ h, __bf16* __restrict__ hb, int n) {
  const int i = blockIdx.x * 256 + threadIdx.x;
  if (i < n) { h[i] = 0.0f; hb[i] = (__bf16)0.0f; }
}

// Per-row logsumexp + argmax: one 256-thread block per row (b*T+t order,
// matching target.reshape(-1)).
__global__ __launch_bounds__(256) void row_stats(
    const float* __restrict__ logits, const int* __restrict__ target,
    float* __restrict__ rowloss, float* __restrict__ rowcorr)
{
  const int row = blockIdx.x;
  const float* x = logits + (size_t)row * kV;
  const int tid = threadIdx.x, lane = tid & 31, wave = tid >> 5;

  float m = -3.4e38f; int mi = 0x7fffffff;
  for (int c = tid; c < kV; c += 256) {
    const float v = x[c];
    if (v > m || (v == m && c < mi)) { m = v; mi = c; }
  }
  #pragma unroll
  for (int o = 16; o > 0; o >>= 1) {
    const float om = __shfl_down(m, o);
    const int   oi = __shfl_down(mi, o);
    if (om > m || (om == m && oi < mi)) { m = om; mi = oi; }
  }
  __shared__ float sm[8]; __shared__ int si[8]; __shared__ float ssum[8];
  if (lane == 0) { sm[wave] = m; si[wave] = mi; }
  __syncthreads();
  if (tid == 0) {
    for (int w = 1; w < 8; ++w)
      if (sm[w] > sm[0] || (sm[w] == sm[0] && si[w] < si[0])) { sm[0] = sm[w]; si[0] = si[w]; }
  }
  __syncthreads();
  m = sm[0];

  float s = 0.0f;
  for (int c = tid; c < kV; c += 256) s += __expf(x[c] - m);
  #pragma unroll
  for (int o = 16; o > 0; o >>= 1) s += __shfl_down(s, o);
  if (lane == 0) ssum[wave] = s;
  __syncthreads();
  if (tid == 0) {
    float tot = 0.0f;
    #pragma unroll
    for (int w = 0; w < 8; ++w) tot += ssum[w];
    const int tgt = target[row];
    rowloss[row] = (__logf(tot) + m) - x[tgt];
    rowcorr[row] = (si[0] == tgt) ? 1.0f : 0.0f;
  }
}

__global__ __launch_bounds__(256) void final_reduce(
    const float* __restrict__ rowloss, const float* __restrict__ rowcorr,
    float* __restrict__ out)
{
  const int tid = threadIdx.x, lane = tid & 31, wave = tid >> 5;
  float ls = 0.0f, cs = 0.0f;
  for (int i = tid; i < kB * kT; i += 256) { ls += rowloss[i]; cs += rowcorr[i]; }
  #pragma unroll
  for (int o = 16; o > 0; o >>= 1) { ls += __shfl_down(ls, o); cs += __shfl_down(cs, o); }
  __shared__ float sa[8], sb[8];
  if (lane == 0) { sa[wave] = ls; sb[wave] = cs; }
  __syncthreads();
  if (tid == 0) {
    float L = 0.0f, C = 0.0f;
    #pragma unroll
    for (int w = 0; w < 8; ++w) { L += sa[w]; C += sb[w]; }
    out[0] = L / (float)(kB * kT);
    out[1] = C / (float)(kB * kT);
  }
}

// ---------------------------------------------------------------------------
// Host driver.
// d_in order (depth-first dict order of setup_inputs):
//   0 enc_emb, 1 dec_emb,
//   cell c in {enc_l0_f,enc_l0_b,enc_l1_f,enc_l1_b,dec_l0_f,dec_l0_b,
//              dec_l1_f,dec_l1_b}: Wih=2+4c, Whh=3+4c, bih=4+4c, bhh=5+4c,
//   34 cls_W, 35 cls_b, 36 source, 37 target.
// ---------------------------------------------------------------------------
extern "C" void kernel_launch(void* const* d_in, const int* in_sizes, int n_in,
                              void* d_out, int out_size, void* d_ws, size_t ws_size,
                              hipStream_t stream) {
  (void)in_sizes; (void)n_in; (void)out_size; (void)ws_size;

  // ---- carve workspace (all offsets 256B aligned) ----
  char* base = (char*)d_ws;
  size_t off = 0;
  auto abf  = [&](size_t n) { __bf16* p = (__bf16*)(base + off);
                              off = (off + n * sizeof(__bf16) + 255) & ~(size_t)255; return p; };
  auto af32 = [&](size_t n) { float* p = (float*)(base + off);
                              off = (off + n * sizeof(float) + 255) & ~(size_t)255; return p; };

  const int kin[8] = {kE, kE, k2H, k2H, kE, kE, k2H, k2H};
  __bf16* wihb[8]; __bf16* whhb[8];
  for (int c = 0; c < 8; ++c) {
    wihb[c] = abf((size_t)kH3 * kin[c]);
    whhb[c] = abf((size_t)kH3 * kH);
  }
  __bf16* clswb = abf((size_t)kV * k2H);
  __bf16* encx  = abf((size_t)kT * kB * kE);
  __bf16* decx  = abf((size_t)kT * kB * kE);
  __bf16* y0    = abf((size_t)kT * kB * k2H);   // encoder layer-0 output
  __bf16* yd0   = abf((size_t)kT * kB * k2H);   // decoder layer-0 output
  __bf16* yd1   = abf((size_t)kT * kB * k2H);   // decoder layer-1 output
  __bf16* hbf   = abf((size_t)4 * kB * kH);     // bf16 hidden mirrors (4 slots)
  float*  hf32  = af32((size_t)4 * kB * kH);    // fp32 hidden (4 slots)
  float*  g     = af32((size_t)2 * kB * kG);    // per-step gate scratch
  float*  rowl  = af32((size_t)kB * kT);
  float*  rowc  = af32((size_t)kB * kT);

  const size_t SL = (size_t)kB * kH;            // slot stride
  float* out = (float*)d_out;

  // ---- weight conversion fp32 -> bf16 (L2-resident afterwards, ~29MB) ----
  for (int c = 0; c < 8; ++c) {
    const int nW = kH3 * kin[c];
    f32_to_bf16_k<<<(nW + 255) / 256, 256, 0, stream>>>((const float*)d_in[2 + 4 * c], wihb[c], nW);
    f32_to_bf16_k<<<(kH3 * kH + 255) / 256, 256, 0, stream>>>((const float*)d_in[3 + 4 * c], whhb[c], kH3 * kH);
  }
  f32_to_bf16_k<<<(kV * k2H + 255) / 256, 256, 0, stream>>>((const float*)d_in[34], clswb, kV * k2H);

  // ---- embeddings (time-major [T,B,E], decoder fused with SOS-shift+relu) ----
  embed_enc<<<(kT * kB * kE) / 256, 256, 0, stream>>>((const float*)d_in[0], (const int*)d_in[36], encx);
  embed_dec<<<(kT * kB * kE) / 256, 256, 0, stream>>>((const float*)d_in[1], (const int*)d_in[37], decx);

  // ---- zero initial hidden state (encoder starts from h=0) ----
  zero_state<<<(4 * kB * kH) / 256, 256, 0, stream>>>(hf32, hbf, 4 * kB * kH);

  // ---- one bi-cell layer = 64 sequential (GEMM, gate) pairs ----
  auto run_layer = [&](const __bf16* xb, int Kin, int cF, int cB,
                       int slotF, int slotB, __bf16* yout, bool reversed) {
    for (int s = 0; s < kT; ++s) {
      const int tF = s;
      const int tB = reversed ? (kT - 1 - s) : s;
      gru_step_gemm<<<dim3(24, 8, 2), 256, 0, stream>>>(
          xb, Kin, wihb[cF], wihb[cB], whhb[cF], whhb[cB],
          hbf + slotF * SL, hbf + slotB * SL, g, tF, tB);
      gru_gate<<<(2 * kB * kH) / 256, 256, 0, stream>>>(
          g,
          (const float*)d_in[4 + 4 * cF], (const float*)d_in[5 + 4 * cF],
          (const float*)d_in[4 + 4 * cB], (const float*)d_in[5 + 4 * cB],
          hf32 + slotF * SL, hf32 + slotB * SL,
          hbf + slotF * SL, hbf + slotB * SL,
          yout, tF, tB);
    }
  };

  // Encoder: layer 0 (reversed bwd scan) -> y0; layer 1 finals only.
  run_layer(encx, kE,  0, 1, 0, 1, y0,      true);
  run_layer(y0,   k2H, 2, 3, 2, 3, nullptr, true);
  // Decoder: all cells run forward in time; h slots already hold encoder
  // finals (slot0/1 = enc l0 f/b, slot2/3 = enc l1 f/b).
  run_layer(decx, kE,  4, 5, 0, 1, yd0, false);
  run_layer(yd0,  k2H, 6, 7, 2, 3, yd1, false);

  // ---- classifier + loss/accuracy ----
  cls_gemm<<<dim3(4, 512), 256, 0, stream>>>(yd1, clswb, (const float*)d_in[35], out + 2);
  row_stats<<<kB * kT, 256, 0, stream>>>(out + 2, (const int*)d_in[37], rowl, rowc);
  final_reduce<<<1, 256, 0, stream>>>(rowl, rowc, out);
}
